// AttentionBlock_22857815949453
// MI455X (gfx1250) — matrix operand order repaired and checked by reading, not compiled, not run on hardware
//
#include <hip/hip_runtime.h>
#include <cstdint>
#include <cstddef>

// Problem constants (B,H,W,C) = (8,64,64,512); HW = 4096.
#define NB 8
#define HW 4096
#define NC 512
#define EPS 1e-3f

typedef __attribute__((ext_vector_type(16))) __bf16 v16bf;
typedef __attribute__((ext_vector_type(8)))  float  v8f;
typedef unsigned short u16;
typedef unsigned int   u32;

// ---------- helpers ----------
__device__ inline u16 f2bf(float f) {           // f32 -> bf16 (RNE)
  u32 u = __float_as_uint(f);
  u32 r = u + 0x7FFFu + ((u >> 16) & 1u);
  return (u16)(r >> 16);
}

union Frag16 { v16bf v; uint4 q[2]; };

// A-operand fragment (16x32 bf16, 16-bit A layout from ISA 7.12.2):
// lane half h: elements 0..7 <- K[8h..8h+7], elements 8..15 <- K[16+8h..23+8h]
__device__ inline v16bf load_fragA(const u16* row, int h) {
  Frag16 u;
  u.q[0] = *(const uint4*)(row + h * 8);
  u.q[1] = *(const uint4*)(row + 16 + h * 8);
  return u.v;
}

// B-operand fragment (32x16 bf16): lane = column n, half h: elements 0..7 <- K[8h..8h+7],
// elements 8..15 <- K[16+8h..23+8h] (same k order as A).
__device__ inline v16bf load_fragB(const u16* col, int h) {
  Frag16 u;
  u.q[0] = *(const uint4*)(col + h * 8);
  u.q[1] = *(const uint4*)(col + 16 + h * 8);
  return u.v;
}

__device__ inline v8f vzero() {
  v8f v;
#pragma unroll
  for (int i = 0; i < 8; ++i) v[i] = 0.f;
  return v;
}

__device__ inline v8f wmma_bf(v16bf a, v16bf b, v8f c) {
  return __builtin_amdgcn_wmma_f32_16x16x32_bf16(false, a, false, b, (short)0, c,
                                                 false, false);
}

// Async global->LDS stage of a contiguous 16*NC bf16 strip (16 KB of traffic,
// 1024 x 16B transfers across 256 threads). Uses CDNA5
// GLOBAL_LOAD_ASYNC_TO_LDS_B128 (no VGPR round trip, tracked by ASYNCcnt).
// LDS byte address = low 32 bits of the generic shared pointer (flat->LDS
// mapping truncates to addr[31:0]).
__device__ inline void stage_strip_async(const u16* __restrict__ gsrc,
                                         u16* lds /* __shared__ */) {
  const uint4* s = (const uint4*)gsrc;
  uint4* d = (uint4*)lds;
#pragma unroll
  for (int k = 0; k < 4; ++k) {
    int i = threadIdx.x + k * 256;           // 0..1023
    u32 loff = (u32)(size_t)(d + i);
    const uint4* gp = s + i;
    asm volatile("global_load_async_to_lds_b128 %0, %1, off"
                 :: "v"(loff), "v"(gp)
                 : "memory");
  }
  asm volatile("s_wait_asynccnt 0" ::: "memory");
}

// ---------- 1a: partial instance-norm stats (coalesced channel-per-thread) ----------
__global__ __launch_bounds__(512) void k_stats_partial(const float* __restrict__ x,
                                                       float* __restrict__ ps) {
  int b = blockIdx.x, chunk = blockIdx.y, c = threadIdx.x;
  const float* xp = x + ((size_t)b * HW + (size_t)chunk * 512) * NC + c;
  float s = 0.f, s2 = 0.f;
  for (int i = 0; i < 512; ++i) { float v = xp[(size_t)i * NC]; s += v; s2 += v * v; }
  float* o = ps + (((size_t)b * 8 + chunk) * NC + c) * 2;
  o[0] = s; o[1] = s2;
}

// ---------- 1b: finalize mean / rstd ----------
__global__ __launch_bounds__(256) void k_stats_final(const float* __restrict__ ps,
                                                     float* __restrict__ st) {
  int idx = blockIdx.x * 256 + threadIdx.x;   // NB*NC = 4096
  int b = idx >> 9, c = idx & 511;
  float s = 0.f, s2 = 0.f;
  for (int j = 0; j < 8; ++j) {
    const float* p = ps + (((size_t)b * 8 + j) * NC + c) * 2;
    s += p[0]; s2 += p[1];
  }
  float mean = s * (1.f / HW);
  float var = s2 * (1.f / HW) - mean * mean;
  st[idx * 2] = mean;
  st[idx * 2 + 1] = rsqrtf(var + EPS);
}

// ---------- 2: normalize -> xn (f32 residual) + bf16 copy ----------
__global__ __launch_bounds__(256) void k_norm(const float* __restrict__ x,
                                              const float* __restrict__ st,
                                              const float* __restrict__ gamma,
                                              const float* __restrict__ beta,
                                              float* __restrict__ xn,
                                              u16* __restrict__ xnb) {
  size_t i = (size_t)blockIdx.x * 256 + threadIdx.x;
  int c = (int)(i & (NC - 1));
  int b = (int)(i >> 21);                    // HW*NC = 2^21
  const float* s = st + ((size_t)b * NC + c) * 2;
  float v = (x[i] - s[0]) * s[1] * gamma[c] + beta[c];
  xn[i] = v;
  xnb[i] = f2bf(v);
}

// ---------- 3: transpose weight to bf16: Wt[cout][cin] ----------
__global__ __launch_bounds__(512) void k_wt(const float* __restrict__ W,
                                            u16* __restrict__ Wt) {
  int co = blockIdx.x, ci = threadIdx.x;
  Wt[(size_t)co * NC + ci] = f2bf(W[(size_t)ci * NC + co]);
}

// ---------- 4: GEMM out = xn @ W + b (W given transposed), bf16 out ----------
// grid.x = NB*256 (b, 16-row tile), grid.y = 4 (128-col slab); 8 waves/block.
__global__ __launch_bounds__(256) void k_gemm_qkv(const u16* __restrict__ xnb,
                                                  const u16* __restrict__ Wt,
                                                  const float* __restrict__ bias,
                                                  u16* __restrict__ out,
                                                  int store_T) {
  int b = blockIdx.x >> 8;
  int mtile = blockIdx.x & 255;
  __shared__ u16 a_lds[16 * NC];
  stage_strip_async(xnb + ((size_t)b * HW + (size_t)mtile * 16) * NC, a_lds);
  __syncthreads();
  int lane = threadIdx.x & 31, wave = threadIdx.x >> 5;
  int h = lane >> 4, nl = lane & 15;
  int n = blockIdx.y * 128 + wave * 16 + nl;
  const u16* brow = Wt + (size_t)n * NC;
  v8f acc = vzero();
#pragma unroll 4
  for (int kc = 0; kc < NC; kc += 32) {
    v16bf a = load_fragA(a_lds + nl * NC + kc, h);
    v16bf bb = load_fragB(brow + kc, h);
    acc = wmma_bf(a, bb, acc);
  }
  float bn = bias[n];
  if (store_T) {                             // V: store transposed [b][c][pos]
    union { uint4 q; u16 t[8]; } tmp;
#pragma unroll
    for (int r = 0; r < 8; ++r) tmp.t[r] = f2bf(acc[r] + bn);
    *(uint4*)(out + ((size_t)b * NC + n) * HW + (size_t)mtile * 16 + h * 8) = tmp.q;
  } else {                                   // Q/K: row-major [b][pos][c]
#pragma unroll
    for (int r = 0; r < 8; ++r) {
      int m = mtile * 16 + h * 8 + r;
      out[((size_t)b * HW + m) * NC + n] = f2bf(acc[r] + bn);
    }
  }
}

// ---------- 5: flash attention: 16 query rows / block, 128-key chunks ----------
__global__ __launch_bounds__(256) void k_attn(const u16* __restrict__ qb,
                                              const u16* __restrict__ kb,
                                              const u16* __restrict__ vbT,
                                              u16* __restrict__ ob) {
  int b = blockIdx.x >> 8;
  int mtile = blockIdx.x & 255;
  __shared__ u16 q_lds[16 * NC];      // 16 KB
  __shared__ float s_lds[16 * 128];   // 8 KB
  __shared__ u16 p_lds[16 * 128];     // 4 KB
  __shared__ float alpha_lds[16];
  __shared__ float rowsum_lds[16];
  stage_strip_async(qb + ((size_t)b * HW + (size_t)mtile * 16) * NC, q_lds);
  int lane = threadIdx.x & 31, wave = threadIdx.x >> 5;
  int h = lane >> 4, nl = lane & 15;
  int srow = threadIdx.x >> 4, sl = threadIdx.x & 15;   // softmax role
  float run_max = -1e30f, run_sum = 0.f;
  v8f acc[4];
#pragma unroll
  for (int t = 0; t < 4; ++t) acc[t] = vzero();
  const float scale = 0.044194173824159216f;            // 512^-0.5
  __syncthreads();

  for (int j = 0; j < HW; j += 128) {
    // prefetch next chunk's K rows / V columns into cache while we compute
    if (j + 128 < HW) {
      const u16* prek = kb + ((size_t)b * HW + (size_t)(j + 128) +
                              (threadIdx.x >> 1)) * NC + (threadIdx.x & 1) * 256;
      __builtin_prefetch(prek, 0, 3);
      const u16* prev = vbT + ((size_t)b * NC + (threadIdx.x << 1)) * HW + j + 128;
      __builtin_prefetch(prev, 0, 3);
    }
    // ---- S tile: this wave handles keys [j + wave*16, +16) ----
    v8f s = vzero();
    {
      const u16* krow = kb + ((size_t)b * HW + (size_t)(j + wave * 16 + nl)) * NC;
#pragma unroll 4
      for (int kc = 0; kc < NC; kc += 32) {
        v16bf a = load_fragA(q_lds + nl * NC + kc, h);
        v16bf bb = load_fragB(krow + kc, h);
        s = wmma_bf(a, bb, s);
      }
    }
#pragma unroll
    for (int r = 0; r < 8; ++r)
      s_lds[(h * 8 + r) * 128 + wave * 16 + nl] = s[r] * scale;
    __syncthreads();

    // ---- online softmax: 16 threads per row, 8 columns each ----
    float vals[8];
    float mloc = -1e30f;
#pragma unroll
    for (int i = 0; i < 8; ++i) {
      vals[i] = s_lds[srow * 128 + sl * 8 + i];
      mloc = fmaxf(mloc, vals[i]);
    }
#pragma unroll
    for (int mk = 8; mk >= 1; mk >>= 1) mloc = fmaxf(mloc, __shfl_xor(mloc, mk, 32));
    float mnew = fmaxf(run_max, mloc);
    float af = __expf(run_max - mnew);
    float ssum = 0.f;
#pragma unroll
    for (int i = 0; i < 8; ++i) {
      float p = __expf(vals[i] - mnew);
      ssum += p;
      p_lds[srow * 128 + sl * 8 + i] = f2bf(p);
    }
#pragma unroll
    for (int mk = 8; mk >= 1; mk >>= 1) ssum += __shfl_xor(ssum, mk, 32);
    run_sum = run_sum * af + ssum;
    run_max = mnew;
    if (sl == 0) alpha_lds[srow] = af;
    __syncthreads();

    // ---- rescale accumulator, then O += P @ V ----
    float al[8];
#pragma unroll
    for (int r = 0; r < 8; ++r) al[r] = alpha_lds[h * 8 + r];
#pragma unroll
    for (int t = 0; t < 4; ++t)
#pragma unroll
      for (int r = 0; r < 8; ++r) acc[t][r] *= al[r];

    for (int kk = 0; kk < 128; kk += 32) {
      v16bf a = load_fragA(p_lds + nl * 128 + kk, h);
#pragma unroll
      for (int t = 0; t < 4; ++t) {
        int n = wave * 64 + t * 16 + nl;
        const u16* vrow = vbT + ((size_t)b * NC + n) * HW + j + kk;
        v16bf bb = load_fragB(vrow, h);
        acc[t] = wmma_bf(a, bb, acc[t]);
      }
    }
    __syncthreads();
  }

  if (sl == 0) rowsum_lds[srow] = run_sum;
  __syncthreads();
  float rinv[8];
#pragma unroll
  for (int r = 0; r < 8; ++r) rinv[r] = 1.f / rowsum_lds[h * 8 + r];
#pragma unroll
  for (int t = 0; t < 4; ++t) {
    int n = wave * 64 + t * 16 + nl;
#pragma unroll
    for (int r = 0; r < 8; ++r) {
      int m = mtile * 16 + h * 8 + r;
      ob[((size_t)b * HW + m) * NC + n] = f2bf(acc[t][r] * rinv[r]);
    }
  }
}

// ---------- 6: out = xn + ob @ Wp + bp ----------
__global__ __launch_bounds__(256) void k_proj(const u16* __restrict__ ob,
                                              const u16* __restrict__ WpT,
                                              const float* __restrict__ bp,
                                              const float* __restrict__ xn,
                                              float* __restrict__ out) {
  int b = blockIdx.x >> 8;
  int mtile = blockIdx.x & 255;
  __shared__ u16 a_lds[16 * NC];
  stage_strip_async(ob + ((size_t)b * HW + (size_t)mtile * 16) * NC, a_lds);
  __syncthreads();
  int lane = threadIdx.x & 31, wave = threadIdx.x >> 5;
  int h = lane >> 4, nl = lane & 15;
  int n = blockIdx.y * 128 + wave * 16 + nl;
  const u16* brow = WpT + (size_t)n * NC;
  v8f acc = vzero();
#pragma unroll 4
  for (int kc = 0; kc < NC; kc += 32) {
    v16bf a = load_fragA(a_lds + nl * NC + kc, h);
    v16bf bb = load_fragB(brow + kc, h);
    acc = wmma_bf(a, bb, acc);
  }
  float bn = bp[n];
#pragma unroll
  for (int r = 0; r < 8; ++r) {
    int m = mtile * 16 + h * 8 + r;
    size_t idx = ((size_t)b * HW + m) * NC + n;
    out[idx] = acc[r] + bn + xn[idx];
  }
}

// ---------- launch ----------
extern "C" void kernel_launch(void* const* d_in, const int* in_sizes, int n_in,
                              void* d_out, int out_size, void* d_ws, size_t ws_size,
                              hipStream_t stream) {
  (void)in_sizes; (void)n_in; (void)out_size; (void)ws_size;
  const float* x     = (const float*)d_in[0];
  const float* gamma = (const float*)d_in[1];
  const float* beta  = (const float*)d_in[2];
  const float* Wq    = (const float*)d_in[3];
  const float* bq    = (const float*)d_in[4];
  const float* Wk    = (const float*)d_in[5];
  const float* bk    = (const float*)d_in[6];
  const float* Wv    = (const float*)d_in[7];
  const float* bv    = (const float*)d_in[8];
  const float* Wp    = (const float*)d_in[9];
  const float* bp    = (const float*)d_in[10];

  // workspace carve (~227 MB total)
  char* base = (char*)d_ws;
  size_t off = 0;
  auto carve = [&](size_t bytes) -> void* {
    void* p = base + off;
    off += (bytes + 255) & ~(size_t)255;
    return p;
  };
  const size_t NTOK = (size_t)NB * HW;              // 32768
  float* ps    = (float*)carve((size_t)NB * 8 * NC * 2 * sizeof(float));
  float* st    = (float*)carve((size_t)NB * NC * 2 * sizeof(float));
  float* xn    = (float*)carve(NTOK * NC * sizeof(float));
  u16*   xnb   = (u16*)  carve(NTOK * NC * sizeof(u16));
  u16*   qb    = (u16*)  carve(NTOK * NC * sizeof(u16));
  u16*   kb    = (u16*)  carve(NTOK * NC * sizeof(u16));
  u16*   vbT   = (u16*)  carve(NTOK * NC * sizeof(u16));
  u16*   obuf  = (u16*)  carve(NTOK * NC * sizeof(u16));
  u16*   WqT   = (u16*)  carve((size_t)NC * NC * sizeof(u16));
  u16*   WkT   = (u16*)  carve((size_t)NC * NC * sizeof(u16));
  u16*   WvT   = (u16*)  carve((size_t)NC * NC * sizeof(u16));
  u16*   WpT   = (u16*)  carve((size_t)NC * NC * sizeof(u16));

  // 1) instance-norm stats
  k_stats_partial<<<dim3(NB, 8), 512, 0, stream>>>(x, ps);
  k_stats_final<<<(NB * NC) / 256, 256, 0, stream>>>(ps, st);
  // 2) normalize + bf16 copy
  k_norm<<<(unsigned)(NTOK * NC / 256), 256, 0, stream>>>(x, st, gamma, beta, xn, xnb);
  // 3) weight transposes to bf16
  k_wt<<<NC, NC, 0, stream>>>(Wq, WqT);
  k_wt<<<NC, NC, 0, stream>>>(Wk, WkT);
  k_wt<<<NC, NC, 0, stream>>>(Wv, WvT);
  k_wt<<<NC, NC, 0, stream>>>(Wp, WpT);
  // 4) QKV projections (V stored transposed)
  dim3 ggrid(NB * (HW / 16), NC / 128);
  k_gemm_qkv<<<ggrid, 256, 0, stream>>>(xnb, WqT, bq, qb, 0);
  k_gemm_qkv<<<ggrid, 256, 0, stream>>>(xnb, WkT, bk, kb, 0);
  k_gemm_qkv<<<ggrid, 256, 0, stream>>>(xnb, WvT, bv, vbT, 1);
  // 5) flash attention
  k_attn<<<NB * (HW / 16), 256, 0, stream>>>(qb, kb, vbT, obuf);
  // 6) projection + bias + residual
  k_proj<<<ggrid, 256, 0, stream>>>(obuf, WpT, bp, xn, (float*)d_out);
}
